// GraphAttentionEmbedding_59511066853416
// MI455X (gfx1250) — matrix operand order repaired and checked by reading, not compiled, not run on hardware
//
#include <hip/hip_runtime.h>
#include <hip/hip_bf16.h>

// ---------------------------------------------------------------------------
// TGAT temporal attention forward, fused for MI455X (gfx1250, wave32, WMMA).
// Heavy GEMMs (K/V proj, Q proj, fc1, fc2) use v_wmma_f32_16x16x32_f16 with
// LDS-staged tiles; weight tiles are staged by the Tensor Data Mover
// (tensor_load_to_lds, padded-stride LDS dest) issued by wave 0 while the
// other waves build the A-tile (fused time2vec cos()). Fragments are
// double-buffered one k-step ahead to decouple ds_load waits from WMMAs.
// Intermediates stored f16 (workload is HBM-bound: ~350 MB @ 23.3 TB/s).
// Workspace use: ~142 MB.
// ---------------------------------------------------------------------------

typedef __attribute__((ext_vector_type(16))) _Float16 v16h;
typedef __attribute__((ext_vector_type(8)))  _Float16 v8h;
typedef __attribute__((ext_vector_type(8)))  float    v8f;
typedef __attribute__((ext_vector_type(4))) unsigned int u32x4;
typedef __attribute__((ext_vector_type(8))) int          i32x8;
typedef __attribute__((ext_vector_type(4))) int          i32x4;

#define BATCH 4096
#define N3    (3 * BATCH)        // 12288 seed nodes
#define NBRS  20
#define DD    128
#define KVROWS ((long)N3 * NBRS) // 245760 (divisible by 64)

// ---------------- WMMA fragment loaders (CDNA5 ISA 7.12.2 layouts) ----------

// A-matrix 16x32 f16. lane L<16: row=L, k=0..7 then 16..23; lane L>=16: row=L-16,
// k=8..15 then 24..31.  Two contiguous 16B LDS loads per fragment.
__device__ inline v16h ld_a_frag(const _Float16* base, int lda, int lane) {
  int row = lane & 15;
  int hi  = (lane >> 4) << 3;          // 0 or 8
  const _Float16* p = base + row * lda + hi;
  v8h lo = *reinterpret_cast<const v8h*>(p);
  v8h up = *reinterpret_cast<const v8h*>(p + 16);
  v16h a;
#pragma unroll
  for (int e = 0; e < 8; ++e) { a[e] = lo[e]; a[e + 8] = up[e]; }
  return a;
}

// B-matrix 32x16 f16, weights staged column-major-in-N ([n][k]) in LDS:
// lane L: column n=L%16; lanes 0-15 hold k=0..7 then 16..23, lanes 16-31 hold k=8..15 then 24..31.
__device__ inline v16h ld_b_frag(const _Float16* base, int ldb, int lane) {
  int col = lane & 15;
  int khi = (lane >> 4) << 3;          // 0 or 8
  const _Float16* p = base + col * ldb + khi;
  v8h lo = *reinterpret_cast<const v8h*>(p);
  v8h up = *reinterpret_cast<const v8h*>(p + 16);
  v16h b;
#pragma unroll
  for (int e = 0; e < 8; ++e) { b[e] = lo[e]; b[e + 8] = up[e]; }
  return b;
}

// ---------------- Weight staging: TDM (async tensor DMA) with fallback ------
// Copy [128][KD] f16 (contiguous, N-major) global -> LDS tile with +8-half
// row padding (stride KD+8). TDM pad_enable reproduces the padding in HW:
//   row bytes = KD*2 -> pad_interval: KD=256 -> 512B = 64x8B -> code 6
//                       KD=128 -> 256B = 32x8B -> code 5
//   pad amount = 16B = 4 DWORDs -> code 3
template <int KD>
__device__ inline void stage_B(const _Float16* __restrict__ WT, _Float16* Blds) {
#if defined(__AMDGCN__) && __has_builtin(__builtin_amdgcn_tensor_load_to_lds)
  if (threadIdx.x < 32) {                       // wave 0 issues the DMA
    const unsigned long long ga = (unsigned long long)(uintptr_t)WT;
    const unsigned lds_addr = (unsigned)(uintptr_t)Blds;  // addr[31:0] = LDS off
    constexpr unsigned PI = (KD == 256) ? 6u : 5u;        // pad interval code
    // D# group0: count=1, lds_addr, 57-bit global addr, type=2 ("image")
    u32x4 g0 = { 1u, lds_addr,
                 (unsigned)(ga & 0xFFFFFFFFull),
                 (unsigned)((ga >> 32) & 0x1FFFFFFull) | (2u << 30) };
    // D# group1: data_size=1(2B), pad_enable, pad_interval, pad_amount=3(4 DW),
    // tensor_dim0=KD, tensor_dim1=128, tile_dim0=KD, tile_dim1=128, stride0=KD
    i32x8 g1 = { (int)((1u << 16) | (1u << 20) | (PI << 22) | (3u << 25)),
                 (int)((unsigned)(KD & 0xFFFF) << 16),
                 (int)(128u << 16),
                 (int)((unsigned)KD << 16),
                 (int)128,
                 (int)KD,
                 0, 0 };
    i32x4 gz = { 0, 0, 0, 0 };                  // 2D tensor: groups 2/3 unused
#if __clang_major__ >= 23
    i32x8 gz8 = { 0, 0, 0, 0, 0, 0, 0, 0 };
    __builtin_amdgcn_tensor_load_to_lds(g0, g1, gz, gz, gz8, 0);
#else
    __builtin_amdgcn_tensor_load_to_lds(g0, g1, gz, gz, 0);
#endif
    __builtin_amdgcn_s_wait_tensorcnt(0);       // TENSORcnt -> 0 before barrier
  }
#else
  constexpr int KC  = KD / 8;
  constexpr int LD8 = (KD + 8) / 8;
  const v8h* s = reinterpret_cast<const v8h*>(WT);
  v8h*       d = reinterpret_cast<v8h*>(Blds);
#pragma unroll
  for (int i = threadIdx.x; i < 128 * KC; i += 256) {
    int n = i / KC, kc = i % KC;
    d[n * LD8 + kc] = s[i];
  }
#endif
}

// ---------------- GEMM core: 64x128 tile, K=KD, 8 waves, pipelined ----------
// Wave w: M-tile = w&3, N-tiles = (w>>2)*4 .. +3. Fragments for step kk+1 are
// loaded before the WMMAs of step kk issue (software pipeline, full unroll).
template <int KD, bool RELU>
__device__ inline void gemm_tile(const _Float16* Alds, const _Float16* Blds,
                                 _Float16* __restrict__ outp,
                                 const float* __restrict__ bias, long row0) {
  constexpr int LDA    = KD + 8;
  constexpr int LDB    = KD + 8;
  constexpr int KSTEPS = KD / 32;
  const int t    = threadIdx.x;
  const int lane = t & 31;
  const int wv   = t >> 5;
  const int mt   = wv & 3;
  const int ng   = wv >> 2;

  const _Float16* Abase = Alds + (mt * 16) * LDA;

  v8f acc[4];
#pragma unroll
  for (int j = 0; j < 4; ++j) {
#pragma unroll
    for (int v = 0; v < 8; ++v) acc[j][v] = 0.0f;
  }

  v16h a_cur = ld_a_frag(Abase, LDA, lane);
  v16h b_cur[4];
#pragma unroll
  for (int j = 0; j < 4; ++j)
    b_cur[j] = ld_b_frag(Blds + ((ng * 4 + j) * 16) * LDB, LDB, lane);

#pragma unroll
  for (int kk = 0; kk < KSTEPS; ++kk) {
    const int kn = (kk + 1 < KSTEPS) ? (kk + 1) : kk;   // clamp (dup last)
    v16h a_nxt = ld_a_frag(Abase + kn * 32, LDA, lane);
    v16h b_nxt[4];
#pragma unroll
    for (int j = 0; j < 4; ++j)
      b_nxt[j] = ld_b_frag(Blds + ((ng * 4 + j) * 16) * LDB + kn * 32, LDB, lane);
#pragma unroll
    for (int j = 0; j < 4; ++j)
      acc[j] = __builtin_amdgcn_wmma_f32_16x16x32_f16(
          false, a_cur, false, b_cur[j], (short)0, acc[j], false, false);
    a_cur = a_nxt;
#pragma unroll
    for (int j = 0; j < 4; ++j) b_cur[j] = b_nxt[j];
  }

  // D layout: lane n = L%16; VGPR v holds m = 8*(L>=16)+v (within 16x16 tile)
#pragma unroll
  for (int j = 0; j < 4; ++j) {
    const int n  = (ng * 4 + j) * 16 + (lane & 15);
    const float bn = bias[n];
#pragma unroll
    for (int v = 0; v < 8; ++v) {
      const int m = mt * 16 + ((lane >> 4) << 3) + v;
      float val = acc[j][v] + bn;
      if (RELU) val = val > 0.0f ? val : 0.0f;
      outp[(row0 + m) * DD + n] = (_Float16)val;
    }
  }
}

// ---------------- Kernel 0: weight transpose + f16 convert ------------------

__global__ __launch_bounds__(256) void prep_weights_kernel(
    const float* __restrict__ Wq, const float* __restrict__ Wk,
    const float* __restrict__ Wv, const float* __restrict__ fc1W,
    const float* __restrict__ fc2W,
    _Float16* __restrict__ WqT, _Float16* __restrict__ WkT,
    _Float16* __restrict__ WvT, _Float16* __restrict__ fc1T,
    _Float16* __restrict__ fc2T) {
  int i = blockIdx.x * 256 + threadIdx.x;
  if (i < 128 * 256) {                    // [n][k] layouts, k-major source
    int n = i >> 8, k = i & 255;
    // Wk/Wv: rows 0..127 multiply zero node feats -> skip; use rows 128..383.
    WkT[i]  = (_Float16)Wk[(128 + k) * DD + n];
    WvT[i]  = (_Float16)Wv[(128 + k) * DD + n];
    WqT[i]  = (_Float16)Wq[k * DD + n];
    fc1T[i] = (_Float16)fc1W[k * DD + n];
  }
  if (i < 128 * 128) {
    int n = i >> 7, k = i & 127;
    fc2T[i] = (_Float16)fc2W[k * DD + n];
  }
}

// ---------------- Kernel 1: fused time2vec + K/V projection -----------------
// A row r = (node,nbr): [ nbr_feats[r] (128) | cos(dt*w+b) (128) ]  -> f16
// grid = (KVROWS/64, 2); blockIdx.y: 0 -> K, 1 -> V.

__global__ __launch_bounds__(256) void kv_proj_kernel(
    const float* __restrict__ nbr_feats, const float* __restrict__ nbr_times,
    const float* __restrict__ time_arr,  const float* __restrict__ t2v_w,
    const float* __restrict__ t2v_b,
    const _Float16* __restrict__ WkT, const _Float16* __restrict__ WvT,
    const float* __restrict__ bk, const float* __restrict__ bv,
    _Float16* __restrict__ Kout, _Float16* __restrict__ Vout) {
  extern __shared__ _Float16 smem[];
  _Float16* Alds = smem;                 // 64 x 264
  _Float16* Blds = smem + 64 * 264;      // 128 x 264

  const bool isV = (blockIdx.y != 0);
  const _Float16* WT   = isV ? WvT : WkT;
  const float*    bias = isV ? bv : bk;
  _Float16*       out  = isV ? Vout : Kout;

  const long row0 = (long)blockIdx.x * 64;
  const int  t    = threadIdx.x;

  // Warm L2/WGP$ for this tile's edge features.
  __builtin_prefetch(nbr_feats + row0 * DD + (long)t * 32, 0, 0);

  stage_B<256>(WT, Blds);                // wave 0: TDM DMA; overlaps A staging

  if (t < 128) {                         // edge-feature half, coalesced
#pragma unroll 4
    for (int r = 0; r < 64; ++r)
      Alds[r * 264 + t] = (_Float16)nbr_feats[(row0 + r) * DD + t];
  } else {                               // time2vec half: cos(dt*w+b)
    const int   c = t - 128;
    const float w = t2v_w[c];
    const float b = t2v_b[c];
#pragma unroll 4
    for (int r = 0; r < 64; ++r) {
      const long rg   = row0 + r;
      const int  node = (int)(rg / NBRS);
      const float dt  = nbr_times[rg] - time_arr[node & (BATCH - 1)];
      Alds[r * 264 + 128 + c] = (_Float16)cosf(dt * w + b);
    }
  }
  __syncthreads();

  gemm_tile<256, false>(Alds, Blds, out, bias, row0);
}

// ---------------- Kernel 2: Q projection (+ persist f16 node feats) ---------

__global__ __launch_bounds__(256) void q_proj_kernel(
    const float* __restrict__ memory, const int* __restrict__ nids,
    const float* __restrict__ t2v_b,
    const _Float16* __restrict__ WqT, const float* __restrict__ bq,
    _Float16* __restrict__ Qout, _Float16* __restrict__ NFout) {
  extern __shared__ _Float16 smem[];
  _Float16* Alds = smem;
  _Float16* Blds = smem + 64 * 264;
  __shared__ int nid_s[64];

  const long row0 = (long)blockIdx.x * 64;
  const int  t    = threadIdx.x;
  if (t < 64) nid_s[t] = nids[row0 + t];

  stage_B<256>(WqT, Blds);
  __syncthreads();                        // nid_s ready

  if (t < 128) {                          // gather memory[nids]
#pragma unroll 4
    for (int r = 0; r < 64; ++r) {
      const _Float16 h = (_Float16)memory[(long)nid_s[r] * DD + t];
      Alds[r * 264 + t]           = h;
      NFout[(row0 + r) * DD + t]  = h;    // reused by fc1 concat
    }
  } else {                                // time2vec(0) = cos(b): constant row
    const int c = t - 128;
    const _Float16 tf = (_Float16)cosf(t2v_b[c]);
#pragma unroll 4
    for (int r = 0; r < 64; ++r) Alds[r * 264 + 128 + c] = tf;
  }
  __syncthreads();

  gemm_tile<256, false>(Alds, Blds, Qout, bq, row0);
}

// ---------------- Kernel 3: masked softmax attention (NBR=20, H=2) ----------

__global__ __launch_bounds__(128) void attn_kernel(
    const _Float16* __restrict__ Q, const _Float16* __restrict__ K,
    const _Float16* __restrict__ V, const int* __restrict__ mask,
    _Float16* __restrict__ Aout) {
  __shared__ float q_s[128];
  __shared__ float sc_s[2][NBRS];
  __shared__ float at_s[2][NBRS];

  const long n = blockIdx.x;
  const int  t = threadIdx.x;

  q_s[t] = (float)Q[n * DD + t];
  __syncthreads();

  if (t < 2 * NBRS) {
    const int h = t / NBRS, m = t % NBRS;
    const _Float16* krow = K + (n * NBRS + m) * DD + h * 64;
    float s = 0.0f;
#pragma unroll
    for (int d = 0; d < 64; ++d) s += q_s[h * 64 + d] * (float)krow[d];
    s *= 0.125f;                          // 1/sqrt(dh=64)
    if (mask[n * NBRS + m] <= 0) s = -1e9f;
    sc_s[h][m] = s;
  }
  __syncthreads();

  if (t < 2) {                            // per-head softmax over 20
    float mx = -3.0e38f;
    for (int m = 0; m < NBRS; ++m) mx = fmaxf(mx, sc_s[t][m]);
    float sum = 0.0f;
    for (int m = 0; m < NBRS; ++m) { float e = __expf(sc_s[t][m] - mx); at_s[t][m] = e; sum += e; }
    const float inv = 1.0f / sum;
    for (int m = 0; m < NBRS; ++m) at_s[t][m] *= inv;
  }
  __syncthreads();

  const int h = t >> 6;
  float o = 0.0f;
#pragma unroll
  for (int m = 0; m < NBRS; ++m)
    o += at_s[h][m] * (float)V[(n * NBRS + m) * DD + t];
  Aout[n * DD + t] = (_Float16)o;
}

// ---------------- Kernel 4: fc1 = relu([attn_out || node_feat] @ W + b) -----

__global__ __launch_bounds__(256) void fc1_kernel(
    const _Float16* __restrict__ A1, const _Float16* __restrict__ A2,
    const _Float16* __restrict__ WT, const float* __restrict__ bias,
    _Float16* __restrict__ out) {
  extern __shared__ _Float16 smem[];
  _Float16* Alds = smem;
  _Float16* Blds = smem + 64 * 264;

  const long row0 = (long)blockIdx.x * 64;
  const int  t    = threadIdx.x;

  stage_B<256>(WT, Blds);

  const _Float16* src = (t < 128) ? A1 : A2;
  const int c = t & 127;
#pragma unroll 4
  for (int r = 0; r < 64; ++r)
    Alds[r * 264 + t] = src[(row0 + r) * DD + c];
  __syncthreads();

  gemm_tile<256, true>(Alds, Blds, out, bias, row0);
}

// ---------------- Kernel 5: fc2 (K=128) -------------------------------------

__global__ __launch_bounds__(256) void fc2_kernel(
    const _Float16* __restrict__ A, const _Float16* __restrict__ WT,
    const float* __restrict__ bias, _Float16* __restrict__ out) {
  extern __shared__ _Float16 smem[];
  _Float16* Alds = smem;                  // 64 x 136
  _Float16* Blds = smem + 64 * 136;       // 128 x 136

  const long row0 = (long)blockIdx.x * 64;
  const int  t    = threadIdx.x;

  stage_B<128>(WT, Blds);
#pragma unroll
  for (int i = t; i < 64 * 128; i += 256) {
    const int r = i >> 7, c = i & 127;
    Alds[r * 136 + c] = A[(row0 + r) * DD + c];
  }
  __syncthreads();

  gemm_tile<128, false>(Alds, Blds, out, bias, row0);
}

// ---------------- Kernel 6: link predictor ----------------------------------
// block b < 4096: pos(src_b, dst_b) -> out[b]; else neg(src_i, neg_i) -> out[4096+i]

__global__ __launch_bounds__(128) void linkpred_kernel(
    const _Float16* __restrict__ Z,
    const float* __restrict__ srcW, const float* __restrict__ srcB,
    const float* __restrict__ dstW, const float* __restrict__ dstB,
    const float* __restrict__ outW, const float* __restrict__ outB,
    float* __restrict__ out) {
  __shared__ float zs_s[128];
  __shared__ float zd_s[128];
  __shared__ float red[128];

  const int  b   = blockIdx.x;
  const int  i   = b & (BATCH - 1);
  const bool neg = (b >= BATCH);
  const long srow = i;                              // z_src
  const long drow = (neg ? 2L * BATCH : BATCH) + i; // z_dst or z_neg
  const int  t   = threadIdx.x;

  zs_s[t] = (float)Z[srow * DD + t];
  zd_s[t] = (float)Z[drow * DD + t];
  __syncthreads();

  float acc = srcB[t] + dstB[t];
#pragma unroll 8
  for (int k = 0; k < 128; ++k)
    acc += zs_s[k] * srcW[k * DD + t] + zd_s[k] * dstW[k * DD + t];
  const float hh = acc > 0.0f ? acc : 0.0f;

  red[t] = hh * outW[t];
  __syncthreads();
#pragma unroll
  for (int s = 64; s > 0; s >>= 1) {
    if (t < s) red[t] += red[t + s];
    __syncthreads();
  }
  if (t == 0) {
    const float x = red[0] + outB[0];
    out[(neg ? BATCH : 0) + i] = 1.0f / (1.0f + __expf(-x));
  }
}

// ---------------------------------------------------------------------------

extern "C" void kernel_launch(void* const* d_in, const int* in_sizes, int n_in,
                              void* d_out, int out_size, void* d_ws, size_t ws_size,
                              hipStream_t stream) {
  const float* memory    = (const float*)d_in[0];
  const int*   nids      = (const int*)  d_in[1];
  const float* time_a    = (const float*)d_in[2];
  const float* nbr_times = (const float*)d_in[3];
  const float* nbr_feats = (const float*)d_in[4];
  const int*   nbr_mask  = (const int*)  d_in[5];
  const float* t2v_w     = (const float*)d_in[6];
  const float* t2v_b     = (const float*)d_in[7];
  const float* Wq   = (const float*)d_in[8];  const float* bq   = (const float*)d_in[9];
  const float* Wk   = (const float*)d_in[10]; const float* bk   = (const float*)d_in[11];
  const float* Wv   = (const float*)d_in[12]; const float* bv   = (const float*)d_in[13];
  const float* fc1W = (const float*)d_in[14]; const float* fc1b = (const float*)d_in[15];
  const float* fc2W = (const float*)d_in[16]; const float* fc2b = (const float*)d_in[17];
  const float* srcW = (const float*)d_in[18]; const float* srcB = (const float*)d_in[19];
  const float* dstW = (const float*)d_in[20]; const float* dstB = (const float*)d_in[21];
  const float* outW = (const float*)d_in[22]; const float* outB = (const float*)d_in[23];

  char*  ws  = (char*)d_ws;
  size_t off = 0;
  auto alloc = [&](size_t bytes) -> void* {
    off = (off + 255) & ~(size_t)255;
    void* p = ws + off;
    off += bytes;
    return p;
  };

  _Float16* WqT  = (_Float16*)alloc(128 * 256 * 2);
  _Float16* WkT  = (_Float16*)alloc(128 * 256 * 2);
  _Float16* WvT  = (_Float16*)alloc(128 * 256 * 2);
  _Float16* fc1T = (_Float16*)alloc(128 * 256 * 2);
  _Float16* fc2T = (_Float16*)alloc(128 * 128 * 2);
  _Float16* Kbuf = (_Float16*)alloc((size_t)KVROWS * DD * 2);   // 62.9 MB
  _Float16* Vbuf = (_Float16*)alloc((size_t)KVROWS * DD * 2);   // 62.9 MB
  _Float16* Qbuf = (_Float16*)alloc((size_t)N3 * DD * 2);
  _Float16* NF16 = (_Float16*)alloc((size_t)N3 * DD * 2);
  _Float16* AO16 = (_Float16*)alloc((size_t)N3 * DD * 2);
  _Float16* H16  = (_Float16*)alloc((size_t)N3 * DD * 2);
  _Float16* Z16  = (_Float16*)alloc((size_t)N3 * DD * 2);
  (void)ws_size; (void)in_sizes; (void)n_in; (void)out_size;

  const size_t lds256 = (size_t)(64 + 128) * 264 * sizeof(_Float16); // ~101 KB
  const size_t lds128 = (size_t)(64 + 128) * 136 * sizeof(_Float16); // ~51 KB

  prep_weights_kernel<<<dim3(128), dim3(256), 0, stream>>>(
      Wq, Wk, Wv, fc1W, fc2W, WqT, WkT, WvT, fc1T, fc2T);

  kv_proj_kernel<<<dim3((unsigned)(KVROWS / 64), 2), dim3(256), lds256, stream>>>(
      nbr_feats, nbr_times, time_a, t2v_w, t2v_b, WkT, WvT, bk, bv, Kbuf, Vbuf);

  q_proj_kernel<<<dim3(N3 / 64), dim3(256), lds256, stream>>>(
      memory, nids, t2v_b, WqT, bq, Qbuf, NF16);

  attn_kernel<<<dim3(N3), dim3(128), 0, stream>>>(
      Qbuf, Kbuf, Vbuf, nbr_mask, AO16);

  fc1_kernel<<<dim3(N3 / 64), dim3(256), lds256, stream>>>(
      AO16, NF16, fc1T, fc1b, H16);

  fc2_kernel<<<dim3(N3 / 64), dim3(256), lds128, stream>>>(
      H16, fc2T, fc2b, Z16);

  linkpred_kernel<<<dim3(2 * BATCH), dim3(128), 0, stream>>>(
      Z16, srcW, srcB, dstW, dstB, outW, outB, (float*)d_out);
}
